// Correlation_43499428773939
// MI455X (gfx1250) — matrix operand order repaired and checked by reading, not compiled, not run on hardware
//
#include <hip/hip_runtime.h>

typedef __attribute__((ext_vector_type(16))) _Float16 v16h;
typedef __attribute__((ext_vector_type(8)))  float    v8f;
typedef __attribute__((ext_vector_type(4)))  unsigned u32x4;
typedef __attribute__((ext_vector_type(4)))  int      i32x4;
typedef __attribute__((ext_vector_type(8)))  int      i32x8;

#define B_    8
#define C_    128
#define H_    128
#define W_    128
#define DISP  4
#define ND    9            // 2*DISP+1 displacements per axis
#define NPAIR (C_ / 2)     // 64 packed channel pairs (x1 side)
#define W1    132          // x1 LDS row stride in dwords (128 + pad)
#define W2F   146          // x2 f32 LDS row stride: 4 halo + 128 data + 14 halo
                           // (16*146 % 64 == 32 -> half-wave rows on disjoint banks)

union Frag { v16h h; unsigned u[8]; };

__device__ __forceinline__ unsigned pack2(float a, float b) {
#if __has_builtin(__builtin_amdgcn_cvt_pkrtz)
    typedef __attribute__((ext_vector_type(2))) __fp16 v2fp16;
    union { v2fp16 h; unsigned u; } p;
    p.h = __builtin_amdgcn_cvt_pkrtz(a, b);   // v_cvt_pk_rtz_f16_f32
    return p.u;
#else
    union { _Float16 f[2]; unsigned u; } p;
    p.f[0] = (_Float16)a;          // even channel -> [15:0]
    p.f[1] = (_Float16)b;          // odd  channel -> [31:16]
    return p.u;
#endif
}

__global__ __launch_bounds__(256)
void corr_wmma_kernel(const float* __restrict__ x1,
                      const float* __restrict__ x2,
                      float* __restrict__ out) {
    __shared__ unsigned lds1[NPAIR * W1];   // x1 row, f16 channel-pair packed
    __shared__ float    lds2f[C_ * W2F];    // x2 row, raw f32, TDM-filled
    __shared__ float    sc[8 * 512];        // per-wave 16x32 f32 D-tile scratch

    const int tid  = threadIdx.x;
    const int lane = tid & 31;
    const int wid  = tid >> 5;
    const int n    = lane & 15;             // N / M index within fragment
    const int hiK  = (lane >> 4) * 8;       // half-wave offset in channel pairs (A: K offset 2*hiK,
                                            // channel-pair rows for B)
    const int h  = blockIdx.x;
    const int b  = blockIdx.y;
    const int w0 = wid * 16;                // this wave's pixel tile

    const float* x1b = x1 + (size_t)b * C_ * H_ * W_;
    const float* x2b = x2 + (size_t)b * C_ * H_ * W_;

    // ---- one-time zero of x2 buffer (halo columns stay zero forever) ----
    for (int i = tid; i < C_ * W2F; i += 256) lds2f[i] = 0.0f;

    // ---- stage x1[b, :, h, :] as packed f16 [cpair][w] ----
    for (int i = tid; i < NPAIR * W_; i += 256) {
        int cp = i >> 7;
        int w  = i & 127;
        const float* p = x1b + ((size_t)(2 * cp) * H_ + h) * W_ + w;
        lds1[cp * W1 + w] = pack2(p[0], p[H_ * W_]);
    }
    __syncthreads();

    // ---- A fragments: x1 tile, M=16 pixels x K=32 channels per chunk ----
    // A layout: lanes0-15 K=0..15, lanes16-31 K=16..31 (matches B)
    Frag a[4];
    #pragma unroll
    for (int kc = 0; kc < 4; ++kc) {
        #pragma unroll
        for (int v = 0; v < 8; ++v) {
            int K  = 2 * v + 2 * hiK;
            int cp = kc * 16 + (K >> 1);
            a[kc].u[v] = lds1[cp * W1 + w0 + n];
        }
    }

    for (int dx = 0; dx < ND; ++dx) {
        __syncthreads();                       // lds2f reuse guard
        const int  hr    = h + dx - DISP;
        const bool rowok = (hr >= 0) && (hr < H_);

        // ---- stage x2[b, :, hr, :] as raw f32 via Tensor Data Mover ----
#ifdef __AMDGCN__
        if (rowok) {
            if (wid == 0) {
                // D# group0: count=1 | lds_addr | 57b global_addr | type=2
                unsigned long long ga =
                    (unsigned long long)(size_t)(x2b + (size_t)hr * W_);
                u32x4 g0 = {};
                g0.x = 1u;                                    // count=1 (valid)
                g0.y = (unsigned)(size_t)(&lds2f[4]);         // data starts at col 4
                g0.z = (unsigned)(ga & 0xFFFFFFFFu);
                g0.w = (unsigned)((ga >> 32) & 0x1FFFFFFu) | (2u << 30); // type=2
                // D# group1: data_size=4B, pad every 128 dwords by 18 dwords,
                // tensor 128x128, tile 128x128, line stride = H*W elements
                i32x8 g1 = {};
                g1[0] = (2 << 16) | (1 << 20) | (6 << 22) | (17 << 25);
                g1[1] = (int)(128u << 16);    // tensor_dim0[15:0]
                g1[2] = (int)(128u << 16);    // tensor_dim1[15:0]
                g1[3] = (int)(128u << 16);    // tile_dim0
                g1[4] = 128;                  // tile_dim1
                g1[5] = H_ * W_;              // tensor_dim0_stride[31:0]
                i32x4 z4 = {};
#if __clang_major__ >= 23
                i32x8 z8 = {};
                __builtin_amdgcn_tensor_load_to_lds(g0, g1, z4, z4, z8, 0);
#else
                __builtin_amdgcn_tensor_load_to_lds(g0, g1, z4, z4, 0);
#endif
                __builtin_amdgcn_s_wait_tensorcnt(0);
            }
        } else {
            for (int i = tid; i < C_ * W_; i += 256) {
                int c = i >> 7, w = i & 127;
                lds2f[c * W2F + 4 + w] = 0.0f;
            }
        }
#else
        for (int i = tid; i < C_ * W_; i += 256) {
            int c = i >> 7, w = i & 127;
            lds2f[c * W2F + 4 + w] =
                rowok ? x2b[((size_t)c * H_ + hr) * W_ + w] : 0.0f;
        }
#endif
        __syncthreads();

        // ---- D[i,j] = sum_c x1[c,w0+i] * x2[c,w0-4+j], j in [0,32) ----
        v8f acc0 = {};   // j = 0..15
        v8f acc1 = {};   // j = 16..31
        #pragma unroll
        for (int kc = 0; kc < 4; ++kc) {
            Frag b0, b1;
            // B layout: lanes0-15 K=0..15, lanes16-31 K=16..31, 2 K per VGPR;
            // convert f32 -> packed f16 at gather time (v_cvt_pk_rtz_f16_f32)
            #pragma unroll
            for (int v = 0; v < 8; ++v) {
                int c0 = 2 * (kc * 16 + hiK + v);
                const float* r0 = lds2f + c0 * W2F;
                b0.u[v] = pack2(r0[w0 + n],      r0[W2F + w0 + n]);
                b1.u[v] = pack2(r0[w0 + 16 + n], r0[W2F + w0 + 16 + n]);
            }
            acc0 = __builtin_amdgcn_wmma_f32_16x16x32_f16(
                       false, a[kc].h, false, b0.h, (short)0, acc0, false, false);
            acc1 = __builtin_amdgcn_wmma_f32_16x16x32_f16(
                       false, a[kc].h, false, b1.h, (short)0, acc1, false, false);
        }

        // ---- dump D tile to per-wave scratch, extract 9 diagonals ----
        float* s = sc + wid * 512;               // 16 rows x 32 cols
        const int M = hiK;                       // 0 or 8
        #pragma unroll
        for (int r = 0; r < 8; ++r) {
            s[(M + r) * 32 + n]      = acc0[r];
            s[(M + r) * 32 + 16 + n] = acc1[r];
        }
        // same-wave LDS ordering; compiler inserts dscnt waits for the reads
        for (int t = lane; t < 144; t += 32) {
            int dy = t >> 4;
            int i  = t & 15;
            float v = s[i * 32 + i + dy] * (1.0f / 128.0f);
            out[(((size_t)b * 81 + (dx * ND + dy)) * H_ + h) * W_ + w0 + i] = v;
        }
    }
}

extern "C" void kernel_launch(void* const* d_in, const int* in_sizes, int n_in,
                              void* d_out, int out_size, void* d_ws, size_t ws_size,
                              hipStream_t stream) {
    const float* x1  = (const float*)d_in[0];
    const float* x2  = (const float*)d_in[1];
    float*       out = (float*)d_out;
    dim3 grid(H_, B_);                 // one workgroup per (h, b) row
    corr_wmma_kernel<<<grid, 256, 0, stream>>>(x1, x2, out);
}
